// Hive_GNN_Model_71356586655930
// MI455X (gfx1250) — matrix operand order repaired and checked by reading, not compiled, not run on hardware
//
#include <hip/hip_runtime.h>
#include <hip/hip_bf16.h>

typedef _Float16 h16v __attribute__((ext_vector_type(16)));
typedef _Float16 h8v  __attribute__((ext_vector_type(8)));
typedef float    f8v  __attribute__((ext_vector_type(8)));

#define N_NODES 100000
#define N_EDGES 1600000
#define N_E2    1700000   // edges + self loops
#define N_MOVES 500000
#define N_GRAPH 64

// ---------- helpers ----------
__device__ __forceinline__ unsigned f2ord(float f) {
  unsigned u = __float_as_uint(f);
  return (u & 0x80000000u) ? ~u : (u | 0x80000000u);
}
__device__ __forceinline__ float ord2f(unsigned u) {
  return (u & 0x80000000u) ? __uint_as_float(u ^ 0x80000000u) : __uint_as_float(~u);
}
__device__ __forceinline__ float elu1(float v) {
  return v > 0.f ? v : (__expf(v) - 1.f);
}
__device__ __forceinline__ float lrelu(float v) { return v > 0.f ? v : 0.2f * v; }
__device__ __forceinline__ h16v mkA(h8v lo, h8v hi) {
  return __builtin_shufflevector(lo, hi, 0, 1, 2, 3, 4, 5, 6, 7, 8, 9, 10, 11, 12, 13, 14, 15);
}
#define WMMA16(a, b, c) __builtin_amdgcn_wmma_f32_16x16x32_f16(false, (a), false, (b), (short)0, (c), false, false)

// ---------- degree + mean self-loop attr ----------
__global__ void k_deg(const int* __restrict__ dst, const float* __restrict__ eattr,
                      float* __restrict__ deg, float* __restrict__ lattr) {
  int e = blockIdx.x * blockDim.x + threadIdx.x;
  if (e >= N_EDGES) return;
  int d = dst[e];
  atomicAdd(&deg[d], 1.f);
#pragma unroll
  for (int k = 0; k < 5; ++k) atomicAdd(&lattr[(size_t)d * 5 + k], eattr[(size_t)e * 5 + k]);
}

__global__ void k_loopdiv(float* __restrict__ lattr, const float* __restrict__ deg) {
  int n = blockIdx.x * blockDim.x + threadIdx.x;
  if (n >= N_NODES) return;
  float dv = deg[n]; dv = dv > 1.f ? dv : 1.f;
#pragma unroll
  for (int k = 0; k < 5; ++k) lattr[(size_t)n * 5 + k] /= dv;
}

// ---------- f32 -> f16 with zero K-padding (layer-1 node features) ----------
__global__ void k_cvtA(const float* __restrict__ src, _Float16* __restrict__ dst,
                       int srcw, int dstw) {
  long i = blockIdx.x * (long)blockDim.x + threadIdx.x;
  long tot = (long)N_NODES * dstw;
  if (i >= tot) return;
  int c = (int)(i % dstw);
  long r = i / dstw;
  dst[i] = (c < srcw) ? (_Float16)src[r * srcw + c] : (_Float16)0.f;
}

// ---------- pack [Wl|Wr] (f32, row stride dout each) into WMMA B-operand layout ----------
// out[((kc*nct + ct)*32 + lane)*16 + e] = B[kc*32 + (lane>=16?8:0) + (e<8?e:e+8)][ct*16 + (lane&15)]
__global__ void k_packB(const float* __restrict__ Wl, const float* __restrict__ Wr,
                        _Float16* __restrict__ out, int din, int dout, int nkc, int nct) {
  int idx = blockIdx.x * blockDim.x + threadIdx.x;
  int tot = nkc * nct * 512;
  if (idx >= tot) return;
  int e = idx & 15;
  int lane = (idx >> 4) & 31;
  int blk = idx >> 9;            // kc*nct + ct
  int ct = blk % nct;
  int kc = blk / nct;
  int n = ct * 16 + (lane & 15);
  int k = kc * 32 + ((lane >= 16) ? 8 : 0) + ((e < 8) ? e : e + 8);
  float v = 0.f;
  if (k < din) v = (n < dout) ? Wl[k * dout + n] : Wr[k * dout + (n - dout)];
  out[idx] = (_Float16)v;
}

// ---------- WMMA GEMM: [N x K](f16) @ packed-B -> split XL/XR (f32); 4 col tiles / wave ----------
__global__ void k_gemm_xlxr(const _Float16* __restrict__ A, const _Float16* __restrict__ Bp,
                            float* __restrict__ XL, float* __restrict__ XR,
                            int K, int dout) {
  int wid  = (blockIdx.x * blockDim.x + threadIdx.x) >> 5;
  int lane = threadIdx.x & 31;
  int ctGroups = (2 * dout) >> 6;                // groups of 4 col tiles
  long totW = (long)(N_NODES / 16) * ctGroups;
  if (wid >= totW) return;                       // wave-uniform
  int rowT = wid / ctGroups, grp = wid % ctGroups;
  int r0 = rowT << 4;
  int ct0 = grp << 2;
  int nkc = K >> 5;
  int nct = (2 * dout) >> 4;
  int lr = lane & 15;
  bool hi = lane >= 16;
  int ka = hi ? 8 : 0;
  const _Float16* arow = A + (size_t)(r0 + lr) * K + ka;
  // per-lane packed-B base; all 4 tiles of a chunk are constant offsets from it
  const _Float16* bptr = Bp + ((size_t)ct0 * 32 + (size_t)lane) * 16;
  size_t bstride = (size_t)nct * 512;            // halfs per K-chunk
  f8v zf = {0.f, 0.f, 0.f, 0.f, 0.f, 0.f, 0.f, 0.f};
  f8v acc[4] = {zf, zf, zf, zf};
  for (int kc = 0; kc < nkc; ++kc) {
    h8v alo = *(const h8v*)(arow);
    h8v ahi = *(const h8v*)(arow + 16);
    h16v b0 = *(const h16v*)(bptr);
    h16v b1 = *(const h16v*)(bptr + 512);
    h16v b2 = *(const h16v*)(bptr + 1024);
    h16v b3 = *(const h16v*)(bptr + 1536);
    h16v a = mkA(alo, ahi);
    acc[0] = WMMA16(a, b0, acc[0]);
    acc[1] = WMMA16(a, b1, acc[1]);
    acc[2] = WMMA16(a, b2, acc[2]);
    acc[3] = WMMA16(a, b3, acc[3]);
    arow += 32;
    bptr += bstride;
  }
  int mb = hi ? 8 : 0;
#pragma unroll
  for (int j = 0; j < 4; ++j) {
    int n0 = (ct0 + j) << 4;
    float* dstp; int cc;
    if (n0 < dout) { dstp = XL; cc = n0 + lr; } else { dstp = XR; cc = n0 + lr - dout; }
#pragma unroll
    for (int i = 0; i < 8; ++i) dstp[(size_t)(r0 + mb + i) * dout + cc] = acc[j][i];
  }
}

// ---------- per-edge attention logit + segment max ----------
__global__ void k_elogit(const int* __restrict__ src, const int* __restrict__ dst,
                         const float* __restrict__ eattr, const float* __restrict__ lattr,
                         const float* __restrict__ xl, const float* __restrict__ xr,
                         const float* __restrict__ We, const float* __restrict__ attw,
                         float* __restrict__ elog, unsigned* __restrict__ mx, int C) {
  long idx = blockIdx.x * (long)blockDim.x + threadIdx.x;
  if (idx >= (long)N_E2 * 4) return;
  int e = (int)(idx >> 2), h = (int)(idx & 3);
  int s, d; const float* ea;
  if (e < N_EDGES) { s = src[e]; d = dst[e]; ea = eattr + (size_t)e * 5; }
  else { s = d = e - N_EDGES; ea = lattr + (size_t)(e - N_EDGES) * 5; }
  float e0 = ea[0], e1 = ea[1], e2 = ea[2], e3 = ea[3], e4 = ea[4];
  int dout = 4 * C;
  const float* xls = xl + (size_t)s * dout + h * C;
  const float* xrd = xr + (size_t)d * dout + h * C;
  const float* w0  = We + h * C;      // We[k*dout + h*C + c]
  const float* at  = attw + h * C;
  float acc = 0.f;
  for (int c = 0; c < C; c += 4) {
    float4 xa = *(const float4*)(xls + c);
    float4 xb = *(const float4*)(xrd + c);
    float4 wv0 = *(const float4*)(w0 + c);
    float4 wv1 = *(const float4*)(w0 + dout + c);
    float4 wv2 = *(const float4*)(w0 + 2 * dout + c);
    float4 wv3 = *(const float4*)(w0 + 3 * dout + c);
    float4 wv4 = *(const float4*)(w0 + 4 * dout + c);
    float4 av = *(const float4*)(at + c);
    float m0 = xa.x + xb.x + e0 * wv0.x + e1 * wv1.x + e2 * wv2.x + e3 * wv3.x + e4 * wv4.x;
    float m1 = xa.y + xb.y + e0 * wv0.y + e1 * wv1.y + e2 * wv2.y + e3 * wv3.y + e4 * wv4.y;
    float m2 = xa.z + xb.z + e0 * wv0.z + e1 * wv1.z + e2 * wv2.z + e3 * wv3.z + e4 * wv4.z;
    float m3 = xa.w + xb.w + e0 * wv0.w + e1 * wv1.w + e2 * wv2.w + e3 * wv3.w + e4 * wv4.w;
    acc = fmaf(lrelu(m0), av.x, acc);
    acc = fmaf(lrelu(m1), av.y, acc);
    acc = fmaf(lrelu(m2), av.z, acc);
    acc = fmaf(lrelu(m3), av.w, acc);
  }
  elog[idx] = acc;
  atomicMax(mx + (size_t)d * 4 + h, f2ord(acc));
}

// ---------- p = exp(logit - max); segment sum ----------
__global__ void k_ep(const int* __restrict__ dst, float* __restrict__ elog,
                     const unsigned* __restrict__ mx, float* __restrict__ z) {
  long idx = blockIdx.x * (long)blockDim.x + threadIdx.x;
  if (idx >= (long)N_E2 * 4) return;
  int e = (int)(idx >> 2), h = (int)(idx & 3);
  int d = (e < N_EDGES) ? dst[e] : e - N_EDGES;
  float p = __expf(elog[idx] - ord2f(mx[(size_t)d * 4 + h]));
  elog[idx] = p;
  atomicAdd(z + (size_t)d * 4 + h, p);
}

// ---------- out[dst] += (p/z) * xl[src] ----------
__global__ void k_eacc(const int* __restrict__ src, const int* __restrict__ dst,
                       const float* __restrict__ elog, const float* __restrict__ z,
                       const float* __restrict__ xl, float* __restrict__ out, int C) {
  long idx = blockIdx.x * (long)blockDim.x + threadIdx.x;
  if (idx >= (long)N_E2 * 4) return;
  int e = (int)(idx >> 2), h = (int)(idx & 3);
  int s, d;
  if (e < N_EDGES) { s = src[e]; d = dst[e]; } else { s = d = e - N_EDGES; }
  float a = elog[idx] / z[(size_t)d * 4 + h];
  int dout = 4 * C;
  const float* xs = xl + (size_t)s * dout + h * C;
  float* od = out + (size_t)d * dout + h * C;
  for (int c = 0; c < C; c += 4) {
    float4 xv = *(const float4*)(xs + c);
    atomicAdd(od + c + 0, a * xv.x);
    atomicAdd(od + c + 1, a * xv.y);
    atomicAdd(od + c + 2, a * xv.z);
    atomicAdd(od + c + 3, a * xv.w);
  }
}

// ---------- concat layers: elu(out + b) -> f16 next-layer input ----------
__global__ void k_fin_concat(const float* __restrict__ out, const float* __restrict__ b,
                             _Float16* __restrict__ hx) {
  long i = blockIdx.x * (long)blockDim.x + threadIdx.x;
  if (i >= (long)N_NODES * 64) return;
  int j = (int)(i & 63);
  float v = elu1(out[i] + b[j]);
  hx[i] = (_Float16)v;
}

// ---------- layer 3: head mean + bias + elu -> h3 (f32) and hx (f16) ----------
__global__ void k_fin_mean(const float* __restrict__ out, const float* __restrict__ b,
                           float* __restrict__ h3, _Float16* __restrict__ hx) {
  long i = blockIdx.x * (long)blockDim.x + threadIdx.x;
  if (i >= (long)N_NODES * 64) return;
  long n = i >> 6; int c = (int)(i & 63);
  const float* o = out + n * 256;
  float v = 0.25f * (o[c] + o[64 + c] + o[128 + c] + o[192 + c]) + b[c];
  v = elu1(v);
  h3[i] = v;
  hx[i] = (_Float16)v;
}

// ---------- fused edge-pair MLP: one wave per 16 moves, all GEMMs via WMMA ----------
__global__ void __launch_bounds__(32)
k_mlp(const _Float16* __restrict__ hx, const int* __restrict__ cm,
      const _Float16* __restrict__ W1p, const float* __restrict__ b1,
      const _Float16* __restrict__ W2p, const float* __restrict__ b2,
      const float* __restrict__ W3, const float* __restrict__ b3,
      float* __restrict__ outv) {
  __shared__ __align__(16) _Float16 z1[16 * 64];
  __shared__ __align__(16) float    z2[16 * 32];
  int t0 = blockIdx.x * 16;
  int lane = threadIdx.x;
  int lr = lane & 15;
  bool hi = lane >= 16;
  int ka = hi ? 8 : 0, mb = hi ? 8 : 0;
  int mi = t0 + lr;
  const _Float16* row0 = hx + (size_t)cm[2 * mi] * 64 + ka;
  const _Float16* row1 = hx + (size_t)cm[2 * mi + 1] * 64 + ka;
  const _Float16* bp1 = W1p + (size_t)lane * 16;   // 16 tiles at constant offsets kc*2048 + j*512
  const _Float16* bp2 = W2p + (size_t)lane * 16;   // 4 tiles at constant offsets

  // z1 = relu(pair @ W1 + b1): 16x128 @ 128x64   (W1 packed: nkc=4, nct=4)
  f8v zf = {0.f, 0.f, 0.f, 0.f, 0.f, 0.f, 0.f, 0.f};
  f8v acc[4] = {zf, zf, zf, zf};
#pragma unroll
  for (int kc = 0; kc < 4; ++kc) {
    const _Float16* srcr = (kc < 2) ? row0 : row1;
    int ko = (kc & 1) << 5;
    h8v alo = *(const h8v*)(srcr + ko);
    h8v ahi = *(const h8v*)(srcr + ko + 16);
    h16v b0 = *(const h16v*)(bp1 + kc * 2048 + 0);
    h16v b1v = *(const h16v*)(bp1 + kc * 2048 + 512);
    h16v b2v = *(const h16v*)(bp1 + kc * 2048 + 1024);
    h16v b3v = *(const h16v*)(bp1 + kc * 2048 + 1536);
    h16v a = mkA(alo, ahi);
    acc[0] = WMMA16(a, b0, acc[0]);
    acc[1] = WMMA16(a, b1v, acc[1]);
    acc[2] = WMMA16(a, b2v, acc[2]);
    acc[3] = WMMA16(a, b3v, acc[3]);
  }
#pragma unroll
  for (int j = 0; j < 4; ++j)
#pragma unroll
    for (int i = 0; i < 8; ++i) {
      int m = mb + i, col = j * 16 + lr;
      float v = acc[j][i] + b1[col];
      v = v > 0.f ? v : 0.f;
      z1[m * 64 + col] = (_Float16)v;
    }
  __syncthreads();

  // z2 = relu(z1 @ W2 + b2): 16x64 @ 64x32   (W2 packed: nkc=2, nct=2)
  f8v a2[2] = {zf, zf};
#pragma unroll
  for (int kc = 0; kc < 2; ++kc) {
    int k0 = kc << 5;
    h8v alo = *(const h8v*)(&z1[lr * 64 + k0 + ka]);
    h8v ahi = *(const h8v*)(&z1[lr * 64 + k0 + ka + 16]);
    h16v b0 = *(const h16v*)(bp2 + kc * 1024 + 0);
    h16v b1v = *(const h16v*)(bp2 + kc * 1024 + 512);
    h16v a = mkA(alo, ahi);
    a2[0] = WMMA16(a, b0, a2[0]);
    a2[1] = WMMA16(a, b1v, a2[1]);
  }
#pragma unroll
  for (int j = 0; j < 2; ++j)
#pragma unroll
    for (int i = 0; i < 8; ++i) {
      int m = mb + i, col = j * 16 + lr;
      float v = a2[j][i] + b2[col];
      z2[m * 32 + col] = v > 0.f ? v : 0.f;
    }
  __syncthreads();

  // logits = z2 @ W3 + b3
  if (lane < 16) {
    float s = b3[0];
#pragma unroll
    for (int c = 0; c < 32; ++c) s = fmaf(z2[lane * 32 + c], W3[c], s);
    outv[t0 + lane] = s;
  }
}

// ---------- global mean pool ----------
__global__ void k_pool(const float* __restrict__ h3, const int* __restrict__ batch,
                       float* __restrict__ g, float* __restrict__ cnt) {
  int n = blockIdx.x * blockDim.x + threadIdx.x;
  if (n >= N_NODES) return;
  int b = batch[n];
  atomicAdd(&cnt[b], 1.f);
  const float* hr = h3 + (size_t)n * 64;
  for (int c = 0; c < 64; ++c) atomicAdd(&g[(size_t)b * 64 + c], hr[c]);
}

// ---------- value head ----------
__global__ void k_value(const float* __restrict__ g, const float* __restrict__ cnt,
                        const float* __restrict__ W1, const float* __restrict__ b1,
                        const float* __restrict__ W2, const float* __restrict__ b2,
                        float* __restrict__ outv) {
  int b = threadIdx.x;
  if (b >= N_GRAPH) return;
  float inv = 1.f / fmaxf(cnt[b], 1.f);
  float acc2 = b2[0];
  for (int j = 0; j < 32; ++j) {
    float v = b1[j];
    for (int c = 0; c < 64; ++c) v = fmaf(g[(size_t)b * 64 + c] * inv, W1[c * 32 + j], v);
    v = v > 0.f ? v : 0.f;
    acc2 = fmaf(v, W2[j], acc2);
  }
  outv[b] = tanhf(acc2);
}

extern "C" void kernel_launch(void* const* d_in, const int* in_sizes, int n_in,
                              void* d_out, int out_size, void* d_ws, size_t ws_size,
                              hipStream_t stream) {
  (void)in_sizes; (void)n_in; (void)out_size; (void)ws_size;
  const float* x     = (const float*)d_in[0];
  const int*   ei    = (const int*)d_in[1];
  const float* eattr = (const float*)d_in[2];
  const int*   cmv   = (const int*)d_in[3];
  const int*   batch = (const int*)d_in[4];
  const float* Wl[3]  = {(const float*)d_in[5],  (const float*)d_in[10], (const float*)d_in[15]};
  const float* Wr[3]  = {(const float*)d_in[6],  (const float*)d_in[11], (const float*)d_in[16]};
  const float* We[3]  = {(const float*)d_in[7],  (const float*)d_in[12], (const float*)d_in[17]};
  const float* att[3] = {(const float*)d_in[8],  (const float*)d_in[13], (const float*)d_in[18]};
  const float* bia[3] = {(const float*)d_in[9],  (const float*)d_in[14], (const float*)d_in[19]};
  const float* epW1 = (const float*)d_in[20]; const float* epb1 = (const float*)d_in[21];
  const float* epW2 = (const float*)d_in[22]; const float* epb2 = (const float*)d_in[23];
  const float* epW3 = (const float*)d_in[24]; const float* epb3 = (const float*)d_in[25];
  const float* vhW1 = (const float*)d_in[26]; const float* vhb1 = (const float*)d_in[27];
  const float* vhW2 = (const float*)d_in[28]; const float* vhb2 = (const float*)d_in[29];
  const int* srcI = ei;
  const int* dstI = ei + N_EDGES;

  size_t off = 0;
  auto alloc = [&](size_t bytes) -> char* {
    char* p = (char*)d_ws + off;
    off += (bytes + 255) & ~(size_t)255;
    return p;
  };
  float*    deg   = (float*)alloc((size_t)N_NODES * 4);
  float*    lattr = (float*)alloc((size_t)N_NODES * 5 * 4);
  char*     mxz   = alloc((size_t)N_NODES * 8 * 4);       // mx (u32 NxH) + z (f32 NxH), contiguous
  unsigned* mx    = (unsigned*)mxz;
  float*    zbuf  = (float*)(mxz + (size_t)N_NODES * 4 * 4);
  float*    h3    = (float*)alloc((size_t)N_NODES * 64 * 4);
  _Float16* hx    = (_Float16*)alloc((size_t)N_NODES * 64 * 2);
  _Float16* Bp    = (_Float16*)alloc((size_t)2 * 32 * 512 * 2);   // packed layer weights (max 64KB)
  _Float16* W1p   = (_Float16*)alloc((size_t)4 * 4 * 512 * 2);
  _Float16* W2p   = (_Float16*)alloc((size_t)2 * 2 * 512 * 2);
  float*    xl    = (float*)alloc((size_t)N_NODES * 256 * 4);
  float*    xr    = (float*)alloc((size_t)N_NODES * 256 * 4);     // reused as out accumulator
  float*    elog  = (float*)alloc((size_t)N_E2 * 4 * 4);
  float*    gpool = (float*)alloc((size_t)(N_GRAPH * 64 + N_GRAPH) * 4);
  float*    cnt   = gpool + N_GRAPH * 64;

  auto cdiv = [](long a, long b) -> unsigned { return (unsigned)((a + b - 1) / b); };

  hipMemsetAsync(deg, 0, (size_t)N_NODES * 4, stream);
  hipMemsetAsync(lattr, 0, (size_t)N_NODES * 5 * 4, stream);
  k_deg<<<cdiv(N_EDGES, 256), 256, 0, stream>>>(dstI, eattr, deg, lattr);
  k_loopdiv<<<cdiv(N_NODES, 256), 256, 0, stream>>>(lattr, deg);
  k_cvtA<<<cdiv((long)N_NODES * 32, 256), 256, 0, stream>>>(x, hx, 14, 32);

  const int KpadA[3] = {32, 64, 64}, dinA[3] = {14, 64, 64}, CcA[3] = {16, 16, 64}, ddA[3] = {64, 64, 256};
  unsigned eb = cdiv((long)N_E2 * 4, 256);
  for (int l = 0; l < 3; ++l) {
    int Kp = KpadA[l], din = dinA[l], C = CcA[l], dout = ddA[l];
    int nkc = Kp >> 5, nct = (2 * dout) >> 4;
    k_packB<<<cdiv(nkc * nct * 512, 256), 256, 0, stream>>>(Wl[l], Wr[l], Bp, din, dout, nkc, nct);
    long waves = (long)(N_NODES / 16) * ((2 * dout) >> 6);
    k_gemm_xlxr<<<cdiv(waves * 32, 256), 256, 0, stream>>>(hx, Bp, xl, xr, Kp, dout);
    hipMemsetAsync(mxz, 0, (size_t)N_NODES * 8 * 4, stream);
    k_elogit<<<eb, 256, 0, stream>>>(srcI, dstI, eattr, lattr, xl, xr, We[l], att[l], elog, mx, C);
    k_ep<<<eb, 256, 0, stream>>>(dstI, elog, mx, zbuf);
    hipMemsetAsync(xr, 0, (size_t)N_NODES * dout * 4, stream);   // xr becomes the output accumulator
    k_eacc<<<eb, 256, 0, stream>>>(srcI, dstI, elog, zbuf, xl, xr, C);
    if (l < 2) k_fin_concat<<<cdiv((long)N_NODES * 64, 256), 256, 0, stream>>>(xr, bia[l], hx);
    else       k_fin_mean  <<<cdiv((long)N_NODES * 64, 256), 256, 0, stream>>>(xr, bia[l], h3, hx);
  }

  k_packB<<<cdiv(4 * 4 * 512, 256), 256, 0, stream>>>(epW1, epW1, W1p, 128, 64, 4, 4);
  k_packB<<<cdiv(2 * 2 * 512, 256), 256, 0, stream>>>(epW2, epW2, W2p, 64, 32, 2, 2);
  k_mlp<<<N_MOVES / 16, 32, 0, stream>>>(hx, cmv, W1p, epb1, W2p, epb2, epW3, epb3, (float*)d_out);
  hipMemsetAsync(gpool, 0, (size_t)(N_GRAPH * 64 + N_GRAPH) * 4, stream);
  k_pool<<<cdiv(N_NODES, 256), 256, 0, stream>>>(h3, batch, gpool, cnt);
  k_value<<<1, 64, 0, stream>>>(gpool, cnt, vhW1, vhb1, vhW2, vhb2, (float*)d_out + N_MOVES);
}
